// Autoencoder_22033182228968
// MI455X (gfx1250) — matrix operand order repaired and checked by reading, not compiled, not run on hardware
//
#include <hip/hip_runtime.h>

// ===========================================================================
// EnCodec-style autoencoder for MI455X (gfx1250, wave32, WMMA + TDM).
// GEMM-shaped math (LSTM projections, RVQ distances, linear layers) runs on
// v_wmma_f32_16x16x32_f16.  Weights are pre-converted once to an f16 image in
// scratch.  A-panels are staged into LDS by the Tensor Data Mover; A
// fragments are hoisted into registers and reused across all N-tiles.
// N-tile loops are kept rolled (unroll 1) with two independent WMMA
// accumulator chains per tile to balance ILP against VGPR pressure.
// ===========================================================================

typedef __attribute__((ext_vector_type(16))) _Float16 v16h;
typedef __attribute__((ext_vector_type(8)))  _Float16 v8h;
typedef __attribute__((ext_vector_type(8)))  float    v8f;
typedef __attribute__((ext_vector_type(4)))  unsigned int u32x4;
typedef __attribute__((ext_vector_type(8)))  int      i32x8;
typedef __attribute__((ext_vector_type(4)))  int      i32x4;

#define NB   128      // batch
#define NT   2048     // input time
#define TL   512      // latent time
#define HID  256
#define GAT  1024     // 4*HID
#define NQ   8
#define BINS 1024
#define MRQ  (NB * TL)

__device__ __forceinline__ float eluf(float x) { return x > 0.f ? x : (__expf(x) - 1.f); }
__device__ __forceinline__ float sigf(float x) { return 1.f / (1.f + __expf(-x)); }

// --------------------------------------------------------------------------
// Tensor Data Mover: stage `rows` x HID f32 tile (row stride strideElems)
// from global into LDS.  Wave 0 issues the DMA, waits on TENSORcnt.
// Caller must __syncthreads() afterwards.
// --------------------------------------------------------------------------
#if __has_builtin(__builtin_amdgcn_tensor_load_to_lds) && __has_builtin(__builtin_amdgcn_s_wait_tensorcnt)
#define HAVE_TDM 1
#else
#define HAVE_TDM 0
#endif

__device__ __forceinline__ void stage_rows(float* dstLds, const float* src,
                                           int rows, int strideElems, int tid) {
#if HAVE_TDM
    if (tid < 32) {
        unsigned long long ga = (unsigned long long)(size_t)src;
        unsigned lds = (unsigned)(size_t)dstLds;       // flat LDS addr, low 32 bits
        u32x4 g0;
        g0[0] = 1u;                                    // count=1, user descriptor
        g0[1] = lds;                                   // lds_addr (bytes)
        g0[2] = (unsigned)(ga & 0xffffffffu);          // global_addr[31:0]
        g0[3] = (unsigned)((ga >> 32) & 0x01ffffffu) | 0x80000000u; // addr[56:32] | type=2
        i32x8 g1;
        g1[0] = (int)(2u << 16);                       // wg_mask=0, data_size=2 (4B)
        g1[1] = (int)(((unsigned)HID & 0xffffu) << 16);        // tensor_dim0 lo
        g1[2] = (int)(((unsigned)HID >> 16) & 0xffffu) | (int)(((unsigned)rows & 0xffffu) << 16);
        g1[3] = (int)(((unsigned)rows >> 16) & 0xffffu) | (int)(((unsigned)HID & 0xffffu) << 16); // tile_dim0
        g1[4] = (int)((unsigned)rows & 0xffffu);       // tile_dim1 (tile_dim2=0)
        g1[5] = (int)(unsigned)strideElems;            // tensor_dim0_stride[31:0]
        g1[6] = 0;
        g1[7] = 0;
        i32x4 z4 = {0, 0, 0, 0};
#if defined(__clang_major__) && __clang_major__ >= 23
        i32x8 z8 = {0, 0, 0, 0, 0, 0, 0, 0};
        __builtin_amdgcn_tensor_load_to_lds(g0, g1, z4, z4, z8, 0);
#else
        __builtin_amdgcn_tensor_load_to_lds(g0, g1, z4, z4, 0);
#endif
        __builtin_amdgcn_s_wait_tensorcnt(0);
    }
#else
    for (int i = tid; i < rows * HID; i += 256)
        dstLds[i] = src[(size_t)(i >> 8) * strideElems + (i & 255)];
#endif
}

// WMMA A operand (16x32 f16) from an LDS f16 row panel: two 16B segments.
__device__ __forceinline__ v16h pack_a_lds(const _Float16* rowBase, int k0, int kb) {
    const v8h* p0 = (const v8h*)(rowBase + k0 + kb);
    const v8h* p1 = (const v8h*)(rowBase + k0 + kb + 16);
    return __builtin_shufflevector(*p0, *p1, 0, 1, 2, 3, 4, 5, 6, 7,
                                   8, 9, 10, 11, 12, 13, 14, 15);
}

// WMMA A operand from an LDS f32 row panel (converted at pack time).
__device__ __forceinline__ v16h pack_a_lds_f32(const float* rowBase, int k0, int kb) {
    v16h a;
#pragma unroll
    for (int i = 0; i < 8; ++i) {
        a[i]     = (_Float16)rowBase[k0 + kb + i];
        a[i + 8] = (_Float16)rowBase[k0 + kb + 16 + i];
    }
    return a;
}

// WMMA B operand (32x16 f16) from the pre-converted global f16 weight image:
// two 8-half segments (k0+8h, k0+16+8h) -> two global_load_b128, no conversions.
__device__ __forceinline__ v16h pack_b_gh(const _Float16* p) {
    const v8h* q = (const v8h*)p;
    return __builtin_shufflevector(q[0], q[2], 0, 1, 2, 3, 4, 5, 6, 7,
                                   8, 9, 10, 11, 12, 13, 14, 15);
}

// --------------------------------------------------------------------------
// elementwise helpers
// --------------------------------------------------------------------------
__global__ void k_zero(float* p, int n) {
    int i = blockIdx.x * 256 + threadIdx.x;
    if (i < n) p[i] = 0.f;
}

__global__ void k_add(const float* __restrict__ a, const float* __restrict__ b,
                      float* __restrict__ o, int n) {
    int i = blockIdx.x * 256 + threadIdx.x;
    if (i < n) o[i] = a[i] + b[i];
}

__global__ void k_tof16(const float* __restrict__ in, _Float16* __restrict__ out, int n) {
    int i = blockIdx.x * 256 + threadIdx.x;
    if (i < n) out[i] = (_Float16)in[i];
}

// --------------------------------------------------------------------------
// direct conv1d, NTC layout: in [B,Tin,Cin], w [Cout,Cin,Kw], out [B,Tout,Cout]
// --------------------------------------------------------------------------
template <int PRE>
__global__ void k_conv(const float* __restrict__ in, const float* __restrict__ w,
                       const float* __restrict__ bias, const float* __restrict__ res,
                       float* __restrict__ out,
                       int Bn, int Tin, int Cin, int Tout, int Cout,
                       int Kw, int stride, int pad) {
    int idx = blockIdx.x * 256 + threadIdx.x;
    int total = Bn * Tout * Cout;
    if (idx >= total) return;
    int co = idx % Cout;
    int t  = (idx / Cout) % Tout;
    int b  = idx / (Cout * Tout);
    float acc = bias[co];
    for (int k = 0; k < Kw; ++k) {
        int ti = t * stride + k - pad;
        if (ti < 0 || ti >= Tin) continue;
        const float* ip = in + ((size_t)b * Tin + ti) * Cin;
        const float* wp = w + (size_t)co * Cin * Kw + k;
        for (int i = 0; i < Cin; ++i) {
            float v = ip[i];
            if (PRE) v = eluf(v);
            acc += v * wp[(size_t)i * Kw];
        }
    }
    if (res) acc += res[idx];
    out[idx] = acc;
}

// --------------------------------------------------------------------------
// ConvTranspose1d (k=3, pad=1, output_padding=stride-1), NTC, pre-ELU input.
// --------------------------------------------------------------------------
__global__ void k_convt(const float* __restrict__ in, const float* __restrict__ w,
                        const float* __restrict__ bias, float* __restrict__ out,
                        int Bn, int Tin, int Cin, int Cout, int stride) {
    int Tout = Tin * stride;
    int idx = blockIdx.x * 256 + threadIdx.x;
    int total = Bn * Tout * Cout;
    if (idx >= total) return;
    int co = idx % Cout;
    int t  = (idx / Cout) % Tout;
    int b  = idx / (Cout * Tout);
    float acc = bias[co];
    for (int k = 0; k < 3; ++k) {
        int d = t + k - 1;
        if (d < 0 || (d % stride) != 0) continue;
        int ti = d / stride;
        if (ti >= Tin) continue;
        const float* ip = in + ((size_t)b * Tin + ti) * Cin;
        const float* wp = w + (size_t)co * Cin * 3 + k;
        for (int i = 0; i < Cin; ++i)
            acc += eluf(ip[i]) * wp[(size_t)i * 3];
    }
    out[idx] = acc;
}

// --------------------------------------------------------------------------
// WMMA GEMM (K = HID):  C[M,N] = act(A) * Bm16[N,256]^T (postELU optional)
// grid = (M/16, N/128), block = 256.  Two independent accumulator chains.
// --------------------------------------------------------------------------
template <int PRE, int POST>
__global__ void __launch_bounds__(256)
k_gemm_bt(const float* __restrict__ A, const _Float16* __restrict__ Bm16,
          float* __restrict__ C, int M, int N) {
    __shared__ float    aS[16 * HID];   // 16 KB TDM destination
    __shared__ _Float16 aH[16 * HID];   //  8 KB activated f16 panel
    const int tid  = threadIdx.x;
    const int lane = tid & 31;
    const int wv   = tid >> 5;
    const int mbase = blockIdx.x * 16;
    const int ncol  = blockIdx.y * 128 + wv * 16;

    stage_rows(aS, A + (size_t)mbase * HID, 16, HID, tid);
    __syncthreads();
    for (int i = tid; i < 16 * HID; i += 256) {
        float v = aS[i];
        if (PRE) v = eluf(v);
        aH[i] = (_Float16)v;
    }
    __syncthreads();

    const int ar = lane & 15;
    const int hf = lane >> 4;
    const int kb = hf * 8;
    const _Float16* Arh = aH + ar * HID;
    const _Float16* Br  = Bm16 + (size_t)(ncol + ar) * HID + hf * 8;
    v8f acc0 = {0.f, 0.f, 0.f, 0.f, 0.f, 0.f, 0.f, 0.f};
    v8f acc1 = acc0;
#pragma unroll
    for (int kk = 0; kk < 8; kk += 2) {
        v16h a0 = pack_a_lds(Arh, kk * 32, kb);
        v16h b0 = pack_b_gh(Br + kk * 32);
        acc0 = __builtin_amdgcn_wmma_f32_16x16x32_f16(false, a0, false, b0,
                                                      (short)0, acc0, false, false);
        v16h a1 = pack_a_lds(Arh, (kk + 1) * 32, kb);
        v16h b1 = pack_b_gh(Br + (kk + 1) * 32);
        acc1 = __builtin_amdgcn_wmma_f32_16x16x32_f16(false, a1, false, b1,
                                                      (short)0, acc1, false, false);
    }
    const int col = ncol + ar;
#pragma unroll
    for (int r = 0; r < 8; ++r) {
        int m = mbase + r + 8 * hf;
        float v = acc0[r] + acc1[r];
        if (POST) v = eluf(v);
        C[(size_t)m * N + col] = v;
    }
}

// --------------------------------------------------------------------------
// LSTM layer (gates i,f,g,o), persistent per-batch-tile kernel (grid = B/16).
// Per step: g[16,1024] = bih+bhh + x_t*wih^T + h*whh^T via WMMA.
// x_t tile TDM-staged each step; A fragments hoisted into registers.
// N-tile loop kept rolled; x-chain and h-chain accumulate independently.
// --------------------------------------------------------------------------
__global__ void __launch_bounds__(256)
k_lstm(const float* __restrict__ X, const _Float16* __restrict__ wih16,
       const _Float16* __restrict__ whh16, const float* __restrict__ bih,
       const float* __restrict__ bhh, float* __restrict__ Y) {
    __shared__ float    xS[16 * HID];   // 16 KB TDM destination
    __shared__ _Float16 hbuf[16 * HID]; //  8 KB
    __shared__ _Float16 gbuf[16 * GAT]; // 32 KB
    const int tid  = threadIdx.x;
    const int lane = tid & 31;
    const int wv   = tid >> 5;
    const int mbase = blockIdx.x * 16;
    const int ar = lane & 15;
    const int hf = lane >> 4;
    const int kb = hf * 8;
    const int grow = tid >> 4;
    const int gcb  = tid & 15;

    for (int i = tid; i < 16 * HID; i += 256) hbuf[i] = (_Float16)0.f;
    float cst[16];
#pragma unroll
    for (int j = 0; j < 16; ++j) cst[j] = 0.f;
    __syncthreads();

    for (int t = 0; t < TL; ++t) {
        stage_rows(xS, X + ((size_t)mbase * TL + t) * HID, 16, TL * HID, tid);
        __syncthreads();
        const float*    xr = xS + ar * HID;
        const _Float16* hr = hbuf + ar * HID;
        v16h ax[8], ah[8];
#pragma unroll
        for (int kk = 0; kk < 8; ++kk) {
            ax[kk] = pack_a_lds_f32(xr, kk * 32, kb);
            ah[kk] = pack_a_lds(hr, kk * 32, kb);
        }
#pragma unroll 1
        for (int nt = 0; nt < 8; ++nt) {
            const int ncol = (wv * 8 + nt) * 16;
            const _Float16* Br = wih16 + (size_t)(ncol + ar) * HID + hf * 8;
            const _Float16* Cr = whh16 + (size_t)(ncol + ar) * HID + hf * 8;
            v8f accx = {0.f, 0.f, 0.f, 0.f, 0.f, 0.f, 0.f, 0.f};
            v8f acch = accx;
#pragma unroll
            for (int kk = 0; kk < 8; ++kk) {   // two independent chains
                v16h bx = pack_b_gh(Br + kk * 32);
                accx = __builtin_amdgcn_wmma_f32_16x16x32_f16(false, ax[kk], false, bx,
                                                              (short)0, accx, false, false);
                v16h bh = pack_b_gh(Cr + kk * 32);
                acch = __builtin_amdgcn_wmma_f32_16x16x32_f16(false, ah[kk], false, bh,
                                                              (short)0, acch, false, false);
            }
            const int col = ncol + ar;
            const float gb = bih[col] + bhh[col];
#pragma unroll
            for (int r = 0; r < 8; ++r)
                gbuf[(r + 8 * hf) * GAT + col] = (_Float16)(accx[r] + acch[r] + gb);
        }
        __syncthreads();
#pragma unroll
        for (int j = 0; j < 16; ++j) {
            const int n = gcb + 16 * j;
            float ig = (float)gbuf[grow * GAT + n];
            float fg = (float)gbuf[grow * GAT + HID + n];
            float gg = (float)gbuf[grow * GAT + 2 * HID + n];
            float og = (float)gbuf[grow * GAT + 3 * HID + n];
            float c = sigf(fg) * cst[j] + sigf(ig) * tanhf(gg);
            cst[j] = c;
            float h = sigf(og) * tanhf(c);
            Y[((size_t)(mbase + grow) * TL + t) * HID + n] = h;
            hbuf[grow * HID + n] = (_Float16)h;
        }
        __syncthreads();
    }
}

// --------------------------------------------------------------------------
// One fused RVQ stage: WMMA scores over all 1024 bins, argmin(|c|^2 - 2 r.c)
// via wave32 shuffles + LDS, then residual/qout update + commitment loss.
// --------------------------------------------------------------------------
__global__ void __launch_bounds__(256)
k_rvq_stage(float* __restrict__ residual, float* __restrict__ qout,
            const _Float16* __restrict__ cb16, const float* __restrict__ cbf,
            const float* __restrict__ norms,
            int* __restrict__ codes, float* __restrict__ lossAcc, int q) {
    __shared__ float    rS[16 * HID];   // 16 KB TDM destination (old residual)
    __shared__ _Float16 rH[16 * HID];   //  8 KB f16 panel
    __shared__ float sv[16][8];
    __shared__ int   si[16][8];
    __shared__ int   fidx[16];
    __shared__ float lred[8];
    const int tid  = threadIdx.x;
    const int lane = tid & 31;
    const int wv   = tid >> 5;
    const int mbase = blockIdx.x * 16;
    const int ar = lane & 15;
    const int hf = lane >> 4;
    const int kb = hf * 8;

    stage_rows(rS, residual + (size_t)mbase * HID, 16, HID, tid);
    __syncthreads();
    for (int i = tid; i < 16 * HID; i += 256) rH[i] = (_Float16)rS[i];
    __syncthreads();

    const _Float16* Arh = rH + ar * HID;
    v16h av[8];
#pragma unroll
    for (int kk = 0; kk < 8; ++kk) av[kk] = pack_a_lds(Arh, kk * 32, kb);

    float bvr[8];
    int   bir[8];
#pragma unroll
    for (int r = 0; r < 8; ++r) { bvr[r] = 3.4e38f; bir[r] = 0; }

#pragma unroll 1
    for (int nt = 0; nt < 8; ++nt) {
        const int ncol = (wv * 8 + nt) * 16;
        const _Float16* Br = cb16 + (size_t)(ncol + ar) * HID + hf * 8;
        v8f acc0 = {0.f, 0.f, 0.f, 0.f, 0.f, 0.f, 0.f, 0.f};
        v8f acc1 = acc0;
#pragma unroll
        for (int kk = 0; kk < 8; kk += 2) {    // two independent chains
            v16h b0 = pack_b_gh(Br + kk * 32);
            acc0 = __builtin_amdgcn_wmma_f32_16x16x32_f16(false, av[kk], false, b0,
                                                          (short)0, acc0, false, false);
            v16h b1 = pack_b_gh(Br + (kk + 1) * 32);
            acc1 = __builtin_amdgcn_wmma_f32_16x16x32_f16(false, av[kk + 1], false, b1,
                                                          (short)0, acc1, false, false);
        }
        const int col = ncol + ar;
        const float nn = norms[col];
#pragma unroll
        for (int r = 0; r < 8; ++r) {
            float mt = nn - 2.f * (acc0[r] + acc1[r]);
            if (mt < bvr[r] || (mt == bvr[r] && col < bir[r])) { bvr[r] = mt; bir[r] = col; }
        }
    }
#pragma unroll
    for (int off = 8; off >= 1; off >>= 1) {
#pragma unroll
        for (int r = 0; r < 8; ++r) {
            float ov = __shfl_xor(bvr[r], off, 32);
            int   oi = __shfl_xor(bir[r], off, 32);
            if (ov < bvr[r] || (ov == bvr[r] && oi < bir[r])) { bvr[r] = ov; bir[r] = oi; }
        }
    }
    if (ar == 0) {
#pragma unroll
        for (int r = 0; r < 8; ++r) { sv[r + 8 * hf][wv] = bvr[r]; si[r + 8 * hf][wv] = bir[r]; }
    }
    __syncthreads();
    if (tid < 16) {
        float bv2 = sv[tid][0];
        int   bi2 = si[tid][0];
        for (int w = 1; w < 8; ++w) {
            float v = sv[tid][w];
            int   iw = si[tid][w];
            if (v < bv2 || (v == bv2 && iw < bi2)) { bv2 = v; bi2 = iw; }
        }
        fidx[tid] = bi2;
        codes[(size_t)(mbase + tid) * NQ + q] = bi2;
    }
    __syncthreads();
    const int urow = tid >> 4;
    const int ucb  = tid & 15;
    const float* qrow = cbf + (size_t)fidx[urow] * HID;   // f32 codebook for update
    const size_t base = (size_t)(mbase + urow) * HID;
    float ll = 0.f;
#pragma unroll
    for (int j = 0; j < 16; ++j) {
        int d = ucb + 16 * j;
        float qv = qrow[d];
        float rv = rS[urow * HID + d] - qv;
        residual[base + d] = rv;
        qout[base + d] += qv;
        ll += rv * rv;
    }
#pragma unroll
    for (int off = 16; off >= 1; off >>= 1) ll += __shfl_xor(ll, off, 32);
    if (lane == 0) lred[wv] = ll;
    __syncthreads();
    if (tid == 0) {
        float s = 0.f;
        for (int w = 0; w < 8; ++w) s += lred[w];
        atomicAdd(lossAcc, s);
    }
}

__global__ void k_cbnorms(const float* __restrict__ cb, float* __restrict__ norms, int n) {
    int i = blockIdx.x * 256 + threadIdx.x;
    if (i >= n) return;
    const float* r = cb + (size_t)i * HID;
    float s = 0.f;
    for (int d = 0; d < HID; ++d) s += r[d] * r[d];
    norms[i] = s;
}

__global__ void k_loss(const float* __restrict__ lossAcc, float* __restrict__ out) {
    if (threadIdx.x == 0 && blockIdx.x == 0) {
        float s = 0.f;
        for (int qq = 0; qq < NQ; ++qq) s += lossAcc[qq];
        out[0] = s / (8.0f * 65536.0f * 256.0f);
    }
}

// ===========================================================================
// Host-side orchestration (JAX pytree alphabetical flatten order, see R0).
// ===========================================================================
extern "C" void kernel_launch(void* const* d_in, const int* in_sizes, int n_in,
                              void* d_out, int out_size, void* d_ws, size_t ws_size,
                              hipStream_t stream) {
    (void)in_sizes; (void)n_in; (void)out_size; (void)ws_size;
    const float* P[48];
    for (int i = 0; i < 48; ++i) P[i] = (const float*)d_in[i];

    float* out_rec  = (float*)d_out;                 // [128,2048,2]
    int*   out_code = (int*)(out_rec + NB * NT * 2); // [128,512,8]
    float* out_loss = out_rec + NB * NT * 2 + NB * TL * NQ;

    float* ws = (float*)d_ws;
    const size_t BIG = 16777216;
    float* b0 = ws;
    float* b1 = ws + BIG;
    float* b2 = ws + 2 * BIG;
    float* b3 = ws + 3 * BIG;
    float* b4 = ws + 4 * BIG;
    float* hid = ws + 5 * BIG;
    float* norms = ws + 6 * BIG;                     // 8192 f32
    float* lossA = norms + 8192;                     // 8 f32
    _Float16* h16 = (_Float16*)(ws + 6 * BIG + 16384);

    const size_t WMAT = 1024 * 256;                  // halfs per LSTM matrix
    _Float16* h_e0ih = h16;              _Float16* h_e0hh = h16 + WMAT;
    _Float16* h_e1ih = h16 + 2 * WMAT;   _Float16* h_e1hh = h16 + 3 * WMAT;
    _Float16* h_d0ih = h16 + 4 * WMAT;   _Float16* h_d0hh = h16 + 5 * WMAT;
    _Float16* h_d1ih = h16 + 6 * WMAT;   _Float16* h_d1hh = h16 + 7 * WMAT;
    _Float16* h_outw = h16 + 8 * WMAT;                       // 65536
    _Float16* h_inw  = h_outw + 65536;                       // 65536
    _Float16* h_cb   = h_inw + 65536;                        // 8*1024*256

    auto tof16 = [&](const float* src, _Float16* dst, int n) {
        k_tof16<<<(n + 255) / 256, 256, 0, stream>>>(src, dst, n);
    };
    tof16(P[41], h_e0ih, WMAT);  tof16(P[40], h_e0hh, WMAT);
    tof16(P[45], h_e1ih, WMAT);  tof16(P[44], h_e1hh, WMAT);
    tof16(P[19], h_d0ih, WMAT);  tof16(P[18], h_d0hh, WMAT);
    tof16(P[23], h_d1ih, WMAT);  tof16(P[22], h_d1hh, WMAT);
    tof16(P[46], h_outw, 65536); tof16(P[15], h_inw, 65536);
    tof16(P[0],  h_cb, NQ * BINS * HID);

    auto conv = [&](const float* in, const float* w, const float* bias, const float* res,
                    float* o, int Tin, int Cin, int Tout, int Cout,
                    int Kw, int stride, int pad, int preElu) {
        int total = NB * Tout * Cout;
        int g = (total + 255) / 256;
        if (preElu)
            k_conv<1><<<g, 256, 0, stream>>>(in, w, bias, res, o, NB, Tin, Cin, Tout, Cout, Kw, stride, pad);
        else
            k_conv<0><<<g, 256, 0, stream>>>(in, w, bias, res, o, NB, Tin, Cin, Tout, Cout, Kw, stride, pad);
    };
    auto convt = [&](const float* in, const float* w, const float* bias, float* o,
                     int Tin, int Cin, int Cout, int stride) {
        int total = NB * Tin * stride * Cout;
        k_convt<<<(total + 255) / 256, 256, 0, stream>>>(in, w, bias, o, NB, Tin, Cin, Cout, stride);
    };
    auto lstm = [&](const float* X, const _Float16* wih, const _Float16* whh,
                    const float* bih, const float* bhh, float* Y) {
        k_lstm<<<NB / 16, 256, 0, stream>>>(X, wih, whh, bih, bhh, Y);
    };

    // ---------------- Encoder ----------------
    conv(P[47], P[37], P[36], nullptr, b0, NT, 2, NT, 64, 3, 1, 1, 0);
    conv(b0, P[28], P[26], nullptr, hid, NT, 64, NT, 32, 3, 1, 1, 1);
    conv(hid, P[29], P[27], b0,     b1,  NT, 32, NT, 64, 1, 1, 0, 1);
    conv(b1, P[25], P[24], nullptr, b0,  NT, 64, NT / 2, 128, 3, 2, 1, 1);
    conv(b0, P[34], P[32], nullptr, hid, NT / 2, 128, NT / 2, 64, 3, 1, 1, 1);
    conv(hid, P[35], P[33], b0,     b1,  NT / 2, 64, NT / 2, 128, 1, 1, 0, 1);
    conv(b1, P[31], P[30], nullptr, b2,  NT / 2, 128, TL, 256, 3, 2, 1, 1);

    lstm(b2, h_e0ih, h_e0hh, P[39], P[38], b3);
    lstm(b3, h_e1ih, h_e1hh, P[43], P[42], b4);
    k_add<<<(MRQ * HID + 255) / 256, 256, 0, stream>>>(b4, b2, b3, MRQ * HID);
    k_gemm_bt<1, 0><<<dim3(MRQ / 16, HID / 128), 256, 0, stream>>>(b3, h_outw, b4, MRQ, HID);

    // ---------------- RVQ ----------------
    k_zero<<<(MRQ * HID + 255) / 256, 256, 0, stream>>>(b2, MRQ * HID);
    k_zero<<<1, 256, 0, stream>>>(lossA, NQ);
    k_cbnorms<<<(NQ * BINS + 255) / 256, 256, 0, stream>>>(P[0], norms, NQ * BINS);
    for (int q = 0; q < NQ; ++q) {
        k_rvq_stage<<<MRQ / 16, 256, 0, stream>>>(b4, b2,
                h_cb + (size_t)q * BINS * HID, P[0] + (size_t)q * BINS * HID,
                norms + q * BINS, out_code, lossA + q, q);
    }
    k_loss<<<1, 32, 0, stream>>>(lossA, out_loss);

    // ---------------- Decoder ----------------
    k_gemm_bt<0, 1><<<dim3(MRQ / 16, HID / 128), 256, 0, stream>>>(b2, h_inw, b3, MRQ, HID);
    lstm(b3, h_d0ih, h_d0hh, P[17], P[16], b4);
    lstm(b4, h_d1ih, h_d1hh, P[21], P[20], b0);
    k_add<<<(MRQ * HID + 255) / 256, 256, 0, stream>>>(b0, b3, b1, MRQ * HID);

    convt(b1, P[6], P[5], b2, TL, 256, 128, 2);
    conv(b2, P[3], P[1], nullptr, hid, NT / 2, 128, NT / 2, 64, 3, 1, 1, 1);
    conv(hid, P[4], P[2], b2,     b3,  NT / 2, 64, NT / 2, 128, 1, 1, 0, 1);
    convt(b3, P[12], P[11], b2, NT / 2, 128, 64, 2);
    conv(b2, P[9], P[7], nullptr, hid, NT, 64, NT, 32, 3, 1, 1, 1);
    conv(hid, P[10], P[8], b2,    b4,  NT, 32, NT, 64, 1, 1, 0, 1);
    conv(b4, P[14], P[13], nullptr, out_rec, NT, 64, NT, 2, 3, 1, 1, 0);
}
